// TT_DLRM_Net_75376676044924
// MI455X (gfx1250) — matrix operand order repaired and checked by reading, not compiled, not run on hardware
//
#include <hip/hip_runtime.h>
#include <hip/hip_bf16.h>

typedef __attribute__((ext_vector_type(16))) _Float16 f16x16;
typedef __attribute__((ext_vector_type(8)))  _Float16 f16x8;
typedef __attribute__((ext_vector_type(8)))  float    f32x8;

#define BATCH 4096

// ---------- WMMA tile loaders (CDNA5 16-bit layouts, wave32) ----------
// A-matrix 16x32 (MxK), row-major source, ld in halves.
// lane<16: row=lane, K = k0+{0..15} contiguous in elems0-15
// lane>=16: row=lane-16, K offsets +16.
__device__ __forceinline__ f16x16 load_a_tile(const _Float16* base, int ld, int k0, int lane) {
  int row  = lane & 15;
  int koff = k0 + ((lane & 16) ? 16 : 0);
  const _Float16* p = base + (size_t)row * ld + koff;
  f16x8 lo = *(const f16x8*)(p);
  f16x8 hi = *(const f16x8*)(p + 8);
  f16x16 r;
#pragma unroll
  for (int e = 0; e < 8; ++e) { r[e] = lo[e]; r[e + 8] = hi[e]; }
  return r;
}

// B-matrix 32x16 (KxN) built from row-major W[N][K] (computes A*W^T).
// lane<16: col n=lane, K = k0+{0..15} contiguous; lane>=16: K = k0+16+{0..15}.
__device__ __forceinline__ f16x16 load_b_tile(const _Float16* W, int ldw, int n0, int k0, int lane) {
  int n  = n0 + (lane & 15);
  int kb = k0 + ((lane & 16) ? 16 : 0);
  const _Float16* p = W + (size_t)n * ldw + kb;
  f16x8 lo = *(const f16x8*)(p);
  f16x8 hi = *(const f16x8*)(p + 8);
  f16x16 r;
#pragma unroll
  for (int e = 0; e < 8; ++e) { r[e] = lo[e]; r[e + 8] = hi[e]; }
  return r;
}

// ---------- Generic f16 WMMA GEMM: D = relu(A[M,K] * W[N,K]^T + bias) ----------
// One wave per 16x16 output tile, 8 waves per block.
__global__ void __launch_bounds__(256) gemm_bias_relu(
    const _Float16* __restrict__ A, const _Float16* __restrict__ W,
    const float* __restrict__ bias, _Float16* __restrict__ D,
    int Mrows, int N, int K, int ldd) {
  int lane = threadIdx.x & 31;
  int wave = threadIdx.x >> 5;
  int tile = blockIdx.x * 8 + wave;
  int ntN  = N >> 4;
  int total = (Mrows >> 4) * ntN;
  if (tile >= total) return;                 // wave-uniform
  int tm = tile / ntN, tn = tile - tm * ntN;
  const _Float16* Ab = A + (size_t)tm * 16 * K;
  const _Float16* Wb = W + (size_t)tn * 16 * K;
  f32x8 acc = {0.f,0.f,0.f,0.f,0.f,0.f,0.f,0.f};
  for (int k0 = 0; k0 < K; k0 += 32) {
    f16x16 av = load_a_tile(Ab, K, k0, lane);
    f16x16 bv = load_b_tile(Wb, K, 0, k0, lane);
    acc = __builtin_amdgcn_wmma_f32_16x16x32_f16(false, av, false, bv, (short)0, acc, false, false);
  }
  int n  = tn * 16 + (lane & 15);
  float bvv = bias[n];
  int rb = (lane & 16) ? 8 : 0;
#pragma unroll
  for (int i = 0; i < 8; ++i) {
    float v = acc[i] + bvv;
    v = v > 0.f ? v : 0.f;
    D[(size_t)(tm * 16 + rb + i) * ldd + n] = (_Float16)v;
  }
}

// ---------- TT embedding lookup: one block per (table, batch element) ----------
// a(4x128) @ b(128x512) -> t(4x512) ; reshape(16x128) @ c(128x4) -> 64 values.
__global__ void __launch_bounds__(256) tt_lookup(
    const int* __restrict__ idx_tt, const float* __restrict__ g0,
    const float* __restrict__ g1, const float* __restrict__ g2,
    _Float16* __restrict__ feats) {
  __shared__ _Float16 Ah[16 * 128];   // A padded to 16 rows (rows 4..15 = 0)
  __shared__ _Float16 Tl[16 * 512];   // phase-1 result (rows 0..3 meaningful)

  int t = blockIdx.x >> 12;
  int b = blockIdx.x & 4095;
  int idx = idx_tt[t * BATCH + b];
  int i0 = idx / 20000;            // // (P1*P2)
  int i1 = (idx / 100) % 200;
  int i2 = idx % 100;
  const float* a_ptr = g0 + ((size_t)t * 200 + i0) * 512;     // (4,128)
  const float* b_ptr = g1 + ((size_t)t * 200 + i1) * 65536;   // (128,512)
  const float* c_ptr = g2 + ((size_t)t * 100 + i2) * 512;     // (128,4)

  // warm L2/WGP$ for the 256KB B core (emits global_prefetch)
  __builtin_prefetch(b_ptr + (size_t)threadIdx.x * 256, 0, 1);

  for (int i = threadIdx.x; i < 16 * 128; i += 256) {
    int r = i >> 7, k = i & 127;
    Ah[i] = (r < 4) ? (_Float16)a_ptr[r * 128 + k] : (_Float16)0.f;
  }
  __syncthreads();

  int lane = threadIdx.x & 31;
  int wave = threadIdx.x >> 5;

  // Phase 1: T(16x512) = Apad(16x128) @ B(128x512); each wave owns 64 columns.
  for (int nt = 0; nt < 4; ++nt) {
    int n0 = wave * 64 + nt * 16;
    f32x8 acc = {0.f,0.f,0.f,0.f,0.f,0.f,0.f,0.f};
    for (int k0 = 0; k0 < 128; k0 += 32) {
      f16x16 av = load_a_tile(Ah, 128, k0, lane);
      int n  = n0 + (lane & 15);
      int kb = k0 + ((lane & 16) ? 16 : 0);
      f16x16 bv;
#pragma unroll
      for (int e = 0; e < 16; ++e) bv[e] = (_Float16)b_ptr[(size_t)(kb + e) * 512 + n];
      acc = __builtin_amdgcn_wmma_f32_16x16x32_f16(false, av, false, bv, (short)0, acc, false, false);
    }
    int n  = n0 + (lane & 15);
    int rb = (lane & 16) ? 8 : 0;
#pragma unroll
    for (int i = 0; i < 8; ++i) Tl[(rb + i) * 512 + n] = (_Float16)acc[i];
  }
  __syncthreads();

  // Phase 2: out(16x4) = T2(16x128) @ c(128x4), N padded to 16. Wave 0 only.
  if (wave == 0) {
    int m = lane & 15;                          // row: q0 = m>>2, q1 = m&3
    int n = lane & 15;                          // col: q2 (valid if <4)
    const _Float16* trow = Tl + (size_t)(m >> 2) * 512 + (m & 3) * 128;
    f32x8 acc = {0.f,0.f,0.f,0.f,0.f,0.f,0.f,0.f};
    for (int k0 = 0; k0 < 128; k0 += 32) {
      int aoff = k0 + ((lane & 16) ? 16 : 0);
      f16x16 av;
#pragma unroll
      for (int e = 0; e < 8; ++e) { av[e] = trow[aoff + e]; av[e + 8] = trow[aoff + 8 + e]; }
      int kb = k0 + ((lane & 16) ? 16 : 0);
      f16x16 bv;
#pragma unroll
      for (int e = 0; e < 16; ++e)
        bv[e] = (n < 4) ? (_Float16)c_ptr[(size_t)(kb + e) * 4 + n] : (_Float16)0.f;
      acc = __builtin_amdgcn_wmma_f32_16x16x32_f16(false, av, false, bv, (short)0, acc, false, false);
    }
    if (n < 4) {
      _Float16* outp = feats + ((size_t)b * 9 + 1 + t) * 64;
      int rb = (lane & 16) ? 8 : 0;
#pragma unroll
      for (int i = 0; i < 8; ++i) outp[(rb + i) * 4 + n] = (_Float16)acc[i];
    }
  }
}

// ---------- std embedding gathers -> feats slots 3..8 ----------
__global__ void __launch_bounds__(256) std_gather(const int* __restrict__ idx_std,
                                                  const float* __restrict__ w_std,
                                                  _Float16* __restrict__ feats) {
  int gid = blockIdx.x * 256 + threadIdx.x;
  if (gid >= 6 * BATCH * 64) return;
  int d = gid & 63;
  int rem = gid >> 6;
  int b = rem & 4095;
  int t = rem >> 12;
  int row = idx_std[t * BATCH + b];
  feats[((size_t)b * 9 + 3 + t) * 64 + d] =
      (_Float16)w_std[((size_t)t * 100000 + row) * 64 + d];
}

// ---------- pairwise interactions -> R = [x(64) | tril(Z)(36) | 0(28)] ----------
__global__ void __launch_bounds__(128) interact(const _Float16* __restrict__ feats,
                                                _Float16* __restrict__ Rh) {
  int b = blockIdx.x;
  int tid = threadIdx.x;
  const _Float16* f = feats + (size_t)b * 576;
  _Float16* r = Rh + (size_t)b * 128;
  if (tid < 64) {
    r[tid] = f[tid];
  } else if (tid < 100) {
    int p = tid - 64;
    int i = 1, acc = 0;
    while (acc + i <= p) { acc += i; ++i; }   // tril_indices(9,-1) order
    int j = p - acc;
    const _Float16* fi = f + i * 64;
    const _Float16* fj = f + j * 64;
    float s = 0.f;
    for (int d = 0; d < 64; ++d) s += (float)fi[d] * (float)fj[d];
    r[tid] = (_Float16)s;
  } else {
    r[tid] = (_Float16)0.f;                   // zero-pad to K=128
  }
}

// ---------- final head: sigmoid(dot256 + b) ----------
__global__ void __launch_bounds__(256) final_head(const _Float16* __restrict__ X,
                                                  const _Float16* __restrict__ w,
                                                  const float* __restrict__ bias,
                                                  float* __restrict__ out) {
  int lane = threadIdx.x & 31;
  int wave = threadIdx.x >> 5;
  int b = blockIdx.x * 8 + wave;
  const _Float16* x = X + (size_t)b * 256;
  float s = 0.f;
  for (int d = lane; d < 256; d += 32) s += (float)x[d] * (float)w[d];
#pragma unroll
  for (int off = 16; off > 0; off >>= 1) s += __shfl_xor(s, off, 32);
  if (lane == 0) out[b] = 1.f / (1.f + expf(-(s + bias[0])));
}

// ---------- f32 -> f16 with zero column padding ----------
__global__ void __launch_bounds__(256) cvt_pad(const float* __restrict__ src,
                                               _Float16* __restrict__ dst,
                                               int rows, int sk, int dk) {
  int gid = blockIdx.x * 256 + threadIdx.x;
  if (gid >= rows * dk) return;
  int rr = gid / dk, c = gid - rr * dk;
  dst[gid] = (c < sk) ? (_Float16)src[(size_t)rr * sk + c] : (_Float16)0.f;
}

extern "C" void kernel_launch(void* const* d_in, const int* in_sizes, int n_in,
                              void* d_out, int out_size, void* d_ws, size_t ws_size,
                              hipStream_t stream) {
  (void)in_sizes; (void)n_in; (void)out_size; (void)ws_size;
  const float* dense_x = (const float*)d_in[0];
  const int*   idx_tt  = (const int*)d_in[1];
  const int*   idx_std = (const int*)d_in[2];
  const float* g0  = (const float*)d_in[3];
  const float* g1  = (const float*)d_in[4];
  const float* g2  = (const float*)d_in[5];
  const float* wst = (const float*)d_in[6];
  const float* bw0 = (const float*)d_in[7];
  const float* bb0 = (const float*)d_in[8];
  const float* bw1 = (const float*)d_in[9];
  const float* bb1 = (const float*)d_in[10];
  const float* bw2 = (const float*)d_in[11];
  const float* bb2 = (const float*)d_in[12];
  const float* tw0 = (const float*)d_in[13];
  const float* tb0 = (const float*)d_in[14];
  const float* tw1 = (const float*)d_in[15];
  const float* tb1 = (const float*)d_in[16];
  const float* tw2 = (const float*)d_in[17];
  const float* tb2 = (const float*)d_in[18];
  float* out = (float*)d_out;

  char* ws = (char*)d_ws;
  size_t o = 0;
  auto alloc = [&](size_t bytes) { size_t r = o; o += (bytes + 255) & ~(size_t)255; return r; };
  _Float16* Xh   = (_Float16*)(ws + alloc((size_t)BATCH * 32 * 2));
  _Float16* W0h  = (_Float16*)(ws + alloc((size_t)512 * 32 * 2));
  _Float16* W1h  = (_Float16*)(ws + alloc((size_t)256 * 512 * 2));
  _Float16* W2h  = (_Float16*)(ws + alloc((size_t)64 * 256 * 2));
  _Float16* TW0h = (_Float16*)(ws + alloc((size_t)512 * 128 * 2));
  _Float16* TW1h = (_Float16*)(ws + alloc((size_t)256 * 512 * 2));
  _Float16* TW2h = (_Float16*)(ws + alloc((size_t)256 * 2));
  _Float16* Y1   = (_Float16*)(ws + alloc((size_t)BATCH * 512 * 2));  // also T1
  _Float16* Y2   = (_Float16*)(ws + alloc((size_t)BATCH * 256 * 2));  // also T2
  _Float16* FE   = (_Float16*)(ws + alloc((size_t)BATCH * 9 * 64 * 2));
  _Float16* Rh   = (_Float16*)(ws + alloc((size_t)BATCH * 128 * 2));

  auto cvt = [&](const float* s, _Float16* d, int rows, int sk, int dk) {
    int total = rows * dk;
    cvt_pad<<<(total + 255) / 256, 256, 0, stream>>>(s, d, rows, sk, dk);
  };
  cvt(dense_x, Xh, BATCH, 13, 32);
  cvt(bw0, W0h, 512, 13, 32);
  cvt(bw1, W1h, 256, 512, 512);
  cvt(bw2, W2h, 64, 256, 256);
  cvt(tw0, TW0h, 512, 100, 128);
  cvt(tw1, TW1h, 256, 512, 512);
  cvt(tw2, TW2h, 1, 256, 256);

  // bottom MLP
  gemm_bias_relu<<<(BATCH / 16) * (512 / 16) / 8, 256, 0, stream>>>(Xh, W0h, bb0, Y1, BATCH, 512, 32, 512);
  gemm_bias_relu<<<(BATCH / 16) * (256 / 16) / 8, 256, 0, stream>>>(Y1, W1h, bb1, Y2, BATCH, 256, 512, 256);
  // last layer writes x directly into feats slot 0 (ldd = 9*64)
  gemm_bias_relu<<<(BATCH / 16) * (64 / 16) / 8, 256, 0, stream>>>(Y2, W2h, bb2, FE, BATCH, 64, 256, 576);

  // embeddings
  std_gather<<<(6 * BATCH * 64 + 255) / 256, 256, 0, stream>>>(idx_std, wst, FE);
  tt_lookup<<<2 * BATCH, 256, 0, stream>>>(idx_tt, g0, g1, g2, FE);

  // interactions -> R
  interact<<<BATCH, 128, 0, stream>>>(FE, Rh);

  // top MLP
  gemm_bias_relu<<<(BATCH / 16) * (512 / 16) / 8, 256, 0, stream>>>(Rh, TW0h, tb0, Y1, BATCH, 512, 128, 512);
  gemm_bias_relu<<<(BATCH / 16) * (256 / 16) / 8, 256, 0, stream>>>(Y1, TW1h, tb1, Y2, BATCH, 256, 512, 256);
  final_head<<<BATCH / 8, 256, 0, stream>>>(Y2, TW2h, tb2, out);
}
